// ResidualBlock_13657996001389
// MI455X (gfx1250) — matrix operand order repaired and checked by reading, not compiled, not run on hardware
//
#include <hip/hip_runtime.h>
#include <hip/hip_bf16.h>

typedef __attribute__((ext_vector_type(16))) __bf16 v16bf;
typedef __attribute__((ext_vector_type(8)))  __bf16 v8bf;
typedef __attribute__((ext_vector_type(8)))  float  v8f;
typedef __attribute__((ext_vector_type(4)))  int    v4i;

// address-space-qualified b128 chunk types for the async-to-LDS builtin
typedef __attribute__((address_space(1))) v4i v4i_g;   // global
typedef __attribute__((address_space(3))) v4i v4i_l;   // LDS

#define N_VOX    200000
#define C_CH     64
#define K_TAPS   27
#define PAD_ROWS (N_VOX + 16)
#define N_TILES  (N_VOX / 16)               // 12500, exact
#define EPS_BN   1e-5f
#define TAP_ELEMS   4096                    // 2 cc * 4 nt * 512 bf16 = 8KB per tap
#define WFRAG_ELEMS (K_TAPS * TAP_ELEMS)    // 110592 bf16 per conv
#define LDS_BYTES   (2 * TAP_ELEMS * 2)     // double-buffered: 16 KB

#if __has_builtin(__builtin_amdgcn_global_load_async_to_lds_b128)
#define HAVE_ASYNC_LDS 1
#else
#define HAVE_ASYNC_LDS 0
#endif

union AFrag { v16bf v; v8bf h[2]; };

// ---- prep: feats f32 -> bf16 padded (zero sentinel rows), zero hpad tail ----
__global__ void prep_feats_kernel(const float* __restrict__ feats,
                                  __bf16* __restrict__ fpad,
                                  __bf16* __restrict__ hpad) {
  int tid = blockIdx.x * blockDim.x + threadIdx.x;
  if (tid >= PAD_ROWS * C_CH) return;
  int row = tid / C_CH;
  if (row < N_VOX) {
    fpad[tid] = (__bf16)feats[tid];
  } else {                 // sentinel rows (gather index N_VOX reads zeros)
    fpad[tid] = (__bf16)0.0f;
    hpad[tid] = (__bf16)0.0f;
  }
}

// ---- prep: W[k][c][n] f32 -> per-lane bf16 B-fragments ----
// Fragment f = (k*2 + cc)*4 + nt ; lane l holds N = nt*16 + (l&15),
// K = cc*32 + (l>=16 ? 16 : 0) + e  for elements e = 0..15 (contiguous).
__global__ void prep_weights_kernel(const float* __restrict__ W,
                                    __bf16* __restrict__ Wfrag) {
  int tid = blockIdx.x * blockDim.x + threadIdx.x;
  if (tid >= WFRAG_ELEMS) return;
  int e    = tid & 15;
  int lane = (tid >> 4) & 31;
  int f    = tid >> 9;
  int nt   = f & 3;
  int cc   = (f >> 2) & 1;
  int k    = f >> 3;
  int c = cc * 32 + ((lane >> 4) << 4) + e;
  int n = nt * 16 + (lane & 15);
  Wfrag[tid] = (__bf16)W[(k * C_CH + c) * C_CH + n];
}

// ---- async copy of one 16B chunk into LDS (gfx1250 async path) ----
__device__ __forceinline__ void async_copy16(const __bf16* g, __bf16* l) {
#if HAVE_ASYNC_LDS
  __builtin_amdgcn_global_load_async_to_lds_b128((v4i_g*)g, (v4i_l*)l,
                                                 /*offset=*/0, /*cpol=*/0);
#else
  *(v8bf*)l = *(const v8bf*)g;           // fallback: load + ds_store
#endif
}

__device__ __forceinline__ void wait_async() {
#if HAVE_ASYNC_LDS
#if __has_builtin(__builtin_amdgcn_s_wait_asynccnt)
  __builtin_amdgcn_s_wait_asynccnt(0);
#else
  asm volatile("s_wait_asynccnt 0x0" ::: "memory");
#endif
#endif
}

// stage one tap's 8 KB of B-fragments into LDS buffer `parity` (whole block)
__device__ __forceinline__ void stage_tap(const __bf16* __restrict__ Wfrag,
                                          __bf16* smem, int k, int parity,
                                          int tid) {
  const __bf16* gsrc = Wfrag + (size_t)k * TAP_ELEMS;
  __bf16* dst = smem + parity * TAP_ELEMS;
  async_copy16(gsrc + tid * 8, dst + tid * 8);                  // chunks 0..255
  async_copy16(gsrc + 2048 + tid * 8, dst + 2048 + tid * 8);    // chunks 256..511
}

// ---- fused conv (+BN, +optional residual) over one 16-voxel strip/wave ----
template <bool RESID>
__global__ void conv_kernel(const __bf16* __restrict__ inpad,
                            const __bf16* __restrict__ Wfrag,
                            const int* __restrict__ nbr,
                            const float* __restrict__ g,
                            const float* __restrict__ bb,
                            const float* __restrict__ mm,
                            const float* __restrict__ vv,
                            const float* __restrict__ feats,
                            float* __restrict__ out_f32,
                            __bf16* __restrict__ out_bf16) {
  extern __shared__ __bf16 smem[];                 // 2 x 8 KB double buffer
  const int tid    = threadIdx.x;
  const int waveId = tid >> 5;
  const int lane   = tid & 31;
  const int tile   = blockIdx.x * 8 + waveId;
  const bool active = (tile < N_TILES);            // wave-uniform

  const int m    = lane & 15;
  const int aoff = (lane >> 4) * 16;               // half-wave column split
  const int base = tile * 16;
  const int* __restrict__ nrow =
      nbr + (size_t)(active ? (base + m) : 0) * K_TAPS;

  v8f acc[4] = {};
  stage_tap(Wfrag, smem, 0, 0, tid);               // prologue: tap 0 -> buf0

  for (int k = 0; k < K_TAPS; ++k) {
    wait_async();                                  // my tap-k chunks in LDS
    __syncthreads();                               // buf[k&1] ready block-wide
    if (k + 1 < K_TAPS)                            // overlap next copy w/ WMMA
      stage_tap(Wfrag, smem, k + 1, (k + 1) & 1, tid);
    if (active) {
      const int row = nrow[k];                     // N_VOX sentinel -> zeros
      const __bf16* __restrict__ fp = inpad + (size_t)row * C_CH;
      const __bf16* wb = smem + (k & 1) * TAP_ELEMS + lane * 16;
#pragma unroll
      for (int cc = 0; cc < 2; ++cc) {             // K chunks [0,32),[32,64)
        AFrag a;
        a.h[0] = *(const v8bf*)(fp + cc * 32 + aoff);
        a.h[1] = *(const v8bf*)(fp + cc * 32 + aoff + 8);
#pragma unroll
        for (int nt = 0; nt < 4; ++nt) {           // A reused across N-tiles
          v16bf b = *(const v16bf*)(wb + (cc * 4 + nt) * 512);  // ds_load
          acc[nt] = __builtin_amdgcn_wmma_f32_16x16x32_bf16(
              false, a.v, false, b, (short)0, acc[nt], false, false);
        }
      }
    }
    __syncthreads();                               // done reading buf[k&1]
  }

  if (!active) return;
  const int ncol  = lane & 15;
  const int rbase = base + ((lane >> 4) << 3);
#pragma unroll
  for (int nt = 0; nt < 4; ++nt) {
    const int n = nt * 16 + ncol;
    const float s = g[n] * rsqrtf(vv[n] + EPS_BN);
    const float o = bb[n] - mm[n] * s;
#pragma unroll
    for (int r = 0; r < 8; ++r) {
      const int idx = (rbase + r) * C_CH + n;
      float val = acc[nt][r] * s + o;
      if (RESID) val += feats[idx];
      val = val > 0.0f ? val : 0.0f;
      if (RESID) out_f32[idx] = val;
      else       out_bf16[idx] = (__bf16)val;
    }
  }
}

extern "C" void kernel_launch(void* const* d_in, const int* in_sizes, int n_in,
                              void* d_out, int out_size, void* d_ws, size_t ws_size,
                              hipStream_t stream) {
  const float* feats = (const float*)d_in[0];
  const int*   nbr   = (const int*)d_in[1];
  const float* W1    = (const float*)d_in[2];
  const float* g1    = (const float*)d_in[3];
  const float* b1    = (const float*)d_in[4];
  const float* m1    = (const float*)d_in[5];
  const float* v1    = (const float*)d_in[6];
  const float* W2    = (const float*)d_in[7];
  const float* g2    = (const float*)d_in[8];
  const float* b2    = (const float*)d_in[9];
  const float* m2    = (const float*)d_in[10];
  const float* v2    = (const float*)d_in[11];
  float* out = (float*)d_out;

  // workspace layout (bf16 staging, all 128B-aligned)
  __bf16* fpad = (__bf16*)d_ws;                    // PAD_ROWS*64
  __bf16* hpad = fpad + (size_t)PAD_ROWS * C_CH;   // PAD_ROWS*64
  __bf16* w1f  = hpad + (size_t)PAD_ROWS * C_CH;   // 110592
  __bf16* w2f  = w1f + WFRAG_ELEMS;                // 110592

  const int prep_threads = PAD_ROWS * C_CH;
  prep_feats_kernel<<<(prep_threads + 255) / 256, 256, 0, stream>>>(feats, fpad, hpad);
  prep_weights_kernel<<<(WFRAG_ELEMS + 255) / 256, 256, 0, stream>>>(W1, w1f);
  prep_weights_kernel<<<(WFRAG_ELEMS + 255) / 256, 256, 0, stream>>>(W2, w2f);

  const int conv_blocks = (N_TILES + 7) / 8;       // 8 waves/block, 1 strip/wave
  conv_kernel<false><<<conv_blocks, 256, LDS_BYTES, stream>>>(
      fpad, w1f, nbr, g1, b1, m1, v1, nullptr, nullptr, hpad);
  conv_kernel<true><<<conv_blocks, 256, LDS_BYTES, stream>>>(
      hpad, w2f, nbr, g2, b2, m2, v2, feats, out, nullptr);
}
